// BidirectionalMultiheadSelfAttention_17901423690272
// MI455X (gfx1250) — compile-verified, matrix operand order checked by reading, not run on hardware
//
#include <hip/hip_runtime.h>
#include <hip/hip_bf16.h>
#include <math.h>
#include <stdint.h>

// Problem constants
constexpr int TT = 1024;   // T
constexpr int BB = 4;      // B
constexpr int EE = 1024;   // E
constexpr int HH = 16;     // H
constexpr int DH = 64;     // Dh
constexpr int SS = 2 * TT; // S

typedef __attribute__((ext_vector_type(16))) _Float16     v16h;
typedef __attribute__((ext_vector_type(8)))  float        v8f;
typedef __attribute__((ext_vector_type(4)))  unsigned int u32x4;
typedef __attribute__((ext_vector_type(8)))  unsigned int u32x8;

__device__ inline v8f wmma_f16(v16h a, v16h b, v8f c) {
  return __builtin_amdgcn_wmma_f32_16x16x32_f16(
      /*neg_a=*/false, a, /*neg_b=*/false, b,
      /*c_mod=*/(short)0, c, /*reuse_a=*/false, /*reuse_b=*/false);
}

// ---- WMMA tile loaders (CDNA5 16-bit A/B VGPR striping, wave32) ----
// Row-major tile: lane holds row (row0 + lane&15); K elements:
//   e<8  -> k0 + (lane>>4)*8 + e
//   e>=8 -> k0 + 16 + (lane>>4)*8 + (e-8)
// Both 8-element runs are contiguous -> two b128 loads per lane (global or ds).
__device__ inline v16h ldtile_row(const _Float16* __restrict__ base,
                                  size_t rstride, int row0, int k0, int lane) {
  const int r  = row0 + (lane & 15);
  const int kh = (lane >> 4) * 8;
  const _Float16* p = base + (size_t)r * rstride + (size_t)(k0 + kh);
  v16h out;
#pragma unroll
  for (int e = 0; e < 8; ++e)  out[e] = p[e];
#pragma unroll
  for (int e = 8; e < 16; ++e) out[e] = p[e + 8];  // 16 + (e-8)
  return out;
}

// A-tile straight from the f32 LDS prob panel, converted to f16.
__device__ inline v16h ldtile_lds_probs(const float* __restrict__ sc, int s0, int lane) {
  const int r  = lane & 15;
  const int kh = (lane >> 4) * 8;
  const float* p = sc + (size_t)r * SS + s0 + kh;
  v16h out;
#pragma unroll
  for (int e = 0; e < 8; ++e)  out[e] = (_Float16)p[e];
#pragma unroll
  for (int e = 8; e < 16; ++e) out[e] = (_Float16)p[e + 8];
  return out;
}

// ---------------- prep: q_in, kv (f32 -> f16) ----------------
__global__ void bmha_prep(const float* __restrict__ fwd, const float* __restrict__ bwd,
                          _Float16* __restrict__ qin, _Float16* __restrict__ kv) {
  const size_t n = (size_t)TT * BB * EE;
  size_t idx = (size_t)blockIdx.x * blockDim.x + threadIdx.x;
  if (idx >= n) return;
  const int e  = (int)(idx % EE);
  const int tb = (int)(idx / EE);
  const int b  = tb % BB;
  const int t  = tb / BB;
  float qv = 0.f;
  if (t > 0)      qv += fwd[(((size_t)(t - 1) * BB + b) * EE) + e];
  if (t < TT - 1) qv += bwd[(((size_t)(t + 1) * BB + b) * EE) + e];
  qin[idx]    = (_Float16)qv;
  kv[idx]     = (_Float16)fwd[idx];
  kv[n + idx] = (_Float16)bwd[idx];
}

// ---------------- weights f32 -> f16 ----------------
__global__ void bmha_convw(const float* __restrict__ wqkv, const float* __restrict__ wo,
                           _Float16* __restrict__ wqkvh, _Float16* __restrict__ woh) {
  size_t i = (size_t)blockIdx.x * blockDim.x + threadIdx.x;
  if (i < (size_t)3 * EE * EE) wqkvh[i] = (_Float16)wqkv[i];
  if (i < (size_t)EE * EE)     woh[i]   = (_Float16)wo[i];
}

__global__ void bmha_zero_f32(float* __restrict__ p, size_t n) {
  size_t i = (size_t)blockIdx.x * blockDim.x + threadIdx.x;
  if (i < n) p[i] = 0.f;
}

// ---------------- generic WMMA GEMM: C[M,N] = A[M,K] @ W[N,K]^T + bias ----------------
// block = 128 threads (4 waves); grid = (N/256, M/16); wave w owns 16x64 output
// (4 independent WMMA accumulators -> A reuse x4, deep XDL pipelining).
// Optional vt output stores V transposed as vT[b][h][d][s] for the PV matmul.
__global__ __launch_bounds__(128) void bmha_gemm_f16(
    const _Float16* __restrict__ A, const _Float16* __restrict__ W,
    const float* __restrict__ bias, _Float16* __restrict__ outh,
    float* __restrict__ outf, _Float16* __restrict__ vt,
    int M, int N, int K) {
  const int wave = threadIdx.x >> 5;
  const int lane = threadIdx.x & 31;
  const int m0 = blockIdx.y * 16;
  const int n0 = blockIdx.x * 256 + wave * 64;

  v8f acc[4] = {{}, {}, {}, {}};
  for (int k0 = 0; k0 < K; k0 += 32) {
    v16h a  = ldtile_row(A, (size_t)K, m0, k0, lane);
    v16h b0 = ldtile_row(W, (size_t)K, n0,      k0, lane);  // W row n == B column n
    v16h b1 = ldtile_row(W, (size_t)K, n0 + 16, k0, lane);
    v16h b2 = ldtile_row(W, (size_t)K, n0 + 32, k0, lane);
    v16h b3 = ldtile_row(W, (size_t)K, n0 + 48, k0, lane);
    acc[0] = wmma_f16(a, b0, acc[0]);
    acc[1] = wmma_f16(a, b1, acc[1]);
    acc[2] = wmma_f16(a, b2, acc[2]);
    acc[3] = wmma_f16(a, b3, acc[3]);
  }

  const int hi = lane >> 4;
#pragma unroll
  for (int c = 0; c < 4; ++c) {
    const int n  = n0 + c * 16 + (lane & 15);
    const float bv = bias[n];
#pragma unroll
    for (int r = 0; r < 8; ++r) {
      const int m = m0 + r + 8 * hi;
      const float val = acc[c][r] + bv;
      if (outf)      outf[(size_t)m * N + n] = val;
      else if (outh) outh[(size_t)m * N + n] = (_Float16)val;
      if (vt) {
        // m = s*BB + b ; n = h*DH + d  ->  vT[((b*HH + h)*DH + d)*SS + s]
        const int bb = m & (BB - 1);
        const int s  = m >> 2;          // m / BB
        const int hh = n >> 6;          // n / DH
        const int d  = n & (DH - 1);
        vt[(((size_t)bb * HH + hh) * DH + d) * SS + s] = (_Float16)val;
      }
    }
  }
}

// ---------------- fused attention: scores -> masked softmax -> avg-probs -> PV ----------------
// grid = (T/16, H, B); block = 128 (4 waves)
// dynamic LDS = 16*S*4 (score panel, 128 KB) + 4 waves * 2 bufs * 2 KB (TDM K staging)
__global__ __launch_bounds__(128) void bmha_attn(
    const _Float16* __restrict__ qh, const _Float16* __restrict__ kh,
    const _Float16* __restrict__ vt, float* __restrict__ avg_out,
    _Float16* __restrict__ attn_h) {
  extern __shared__ float sc[];  // [16][SS] score panel, then K staging
  const int t0 = blockIdx.x * 16;
  const int h  = blockIdx.y;
  const int b  = blockIdx.z;
  const int tid  = threadIdx.x;
  const int wave = tid >> 5;
  const int lane = tid & 31;
  const size_t rs = (size_t)BB * EE;  // row stride of (row, b, e) activations

  const _Float16* qbase  = qh + (size_t)b * EE + (size_t)h * DH;
  const _Float16* kbase  = kh + (size_t)b * EE + (size_t)h * DH;
  const _Float16* vtbase = vt + ((size_t)b * HH + h) * DH * SS;  // [d][s]

  // --- Phase 1: S_panel[16][SS] = Q_tile @ K^T, K tiles DMAed into LDS by the
  //     Tensor Data Mover, double-buffered, synchronized with s_wait_tensorcnt ---
  {
    v16h a0 = ldtile_row(qbase, rs, t0, 0, lane);
    v16h a1 = ldtile_row(qbase, rs, t0, 32, lane);

    const int wv = __builtin_amdgcn_readfirstlane(wave);
    // Per-wave staging: two 16x64 f16 tiles (2 KB each) after the score panel.
    _Float16* stage = (_Float16*)sc + (size_t)16 * SS * 2 + (size_t)wv * 2048;

    // D# group1 (constant): data_size=2B, tensor_dim0=64, tensor_dim1=16,
    // tile_dim0=64, tile_dim1=16, tensor_dim0_stride=B*E elements.
    const u32x8 g1 = { 0x00010000u,            // data_size=1 (2 bytes)
                       64u << 16,              // tensor_dim0 = 64 (bits 79:48 lo)
                       16u << 16,              // tensor_dim1 = 16 (bits 111:80 lo)
                       64u << 16,              // tile_dim0 = 64 (bits 127:112)
                       16u,                    // tile_dim1 = 16
                       (unsigned)(BB * EE),    // tensor_dim0_stride = 4096
                       0u, 0u };

    auto tdm_issue = [&](int s0, int buf) {
      const unsigned long long ga =
          (unsigned long long)(uintptr_t)(kbase + (size_t)s0 * rs);
      // Generic LDS address keeps the LDS byte offset in addr[31:0].
      const unsigned lds_off =
          (unsigned)(unsigned long long)(uintptr_t)(void*)(stage + buf * 1024);
      const u32x4 g0 = { 1u,                          // count=1, user descriptor
                         lds_off,                     // lds_addr
                         (unsigned)ga,                // global_addr[31:0]
                         (unsigned)((ga >> 32) & 0x1FFFFFFull) | (2u << 30) };
      asm volatile("tensor_load_to_lds %0, %1" : : "s"(g0), "s"(g1) : "memory");
    };

    tdm_issue(wv * 16, 0);
    int buf = 0;
    for (int s0 = wv * 16; s0 < SS; s0 += 64) {
      if (s0 + 64 < SS) {
        tdm_issue(s0 + 64, buf ^ 1);
        __builtin_amdgcn_s_wait_tensorcnt(1);  // current buffer complete (in-order)
      } else {
        __builtin_amdgcn_s_wait_tensorcnt(0);
      }
      const _Float16* sb = stage + buf * 1024;  // 16 rows x 64 cols, row stride 64
      v8f acc = {};
      v16h b0 = ldtile_row(sb, 64, 0, 0, lane);   // staged K row s == B column s
      acc = wmma_f16(a0, b0, acc);
      v16h b1 = ldtile_row(sb, 64, 0, 32, lane);
      acc = wmma_f16(a1, b1, acc);
      const int n = lane & 15, hi = lane >> 4;
#pragma unroll
      for (int r = 0; r < 8; ++r)
        sc[(size_t)(r + 8 * hi) * SS + s0 + n] = acc[r];
      buf ^= 1;
    }
  }
  __syncthreads();

  // --- Phase 2: masked softmax, f32; 8 lanes per row; accumulate head-average ---
  {
    const int m = tid >> 3;        // row 0..15
    const int g = tid & 7;         // lane group within row
    const int t = t0 + m;
    float* row = sc + (size_t)m * SS;

    float mx = -INFINITY;
    for (int j = g; j < SS; j += 8) {
      const bool ok = (j <= t - 1) || (j >= t + TT + 1);
      const float x = ok ? row[j] : -INFINITY;
      mx = fmaxf(mx, x);
    }
    mx = fmaxf(mx, __shfl_xor(mx, 1));
    mx = fmaxf(mx, __shfl_xor(mx, 2));
    mx = fmaxf(mx, __shfl_xor(mx, 4));

    float sum = 0.f;
    for (int j = g; j < SS; j += 8) {
      const bool ok = (j <= t - 1) || (j >= t + TT + 1);
      const float x = ok ? __expf(row[j] - mx) : 0.f;
      row[j] = x;
      sum += x;
    }
    sum += __shfl_xor(sum, 1);
    sum += __shfl_xor(sum, 2);
    sum += __shfl_xor(sum, 4);
    const float inv = 1.f / sum;

    float* avg = avg_out + ((size_t)b * TT + t) * SS;
    for (int j = g; j < SS; j += 8) {
      const float p = row[j] * inv;
      row[j] = p;
      atomicAdd(&avg[j], p * (1.f / HH));
    }
  }
  __syncthreads();

  // --- Phase 3: attn_tile[16][64] = P[16][SS] @ V[SS][64]; wave owns 16-col d-tile.
  //     V is pre-transposed (vT[d][s]) so B-tiles are contiguous row loads. ---
  {
    const int d0 = wave * 16;
    v8f acc = {};
    for (int s0 = 0; s0 < SS; s0 += 32) {
      v16h ap = ldtile_lds_probs(sc, s0, lane);
      v16h bv = ldtile_row(vtbase, (size_t)SS, d0, s0, lane);  // vT row d == B column d
      acc = wmma_f16(ap, bv, acc);
    }
    const int n = lane & 15, hi = lane >> 4;
#pragma unroll
    for (int r = 0; r < 8; ++r) {
      const int t = t0 + r + 8 * hi;
      attn_h[(((size_t)t * BB + b) * EE) + (size_t)h * DH + d0 + n] = (_Float16)acc[r];
    }
  }
}

extern "C" void kernel_launch(void* const* d_in, const int* in_sizes, int n_in,
                              void* d_out, int out_size, void* d_ws, size_t ws_size,
                              hipStream_t stream) {
  const float* fwd_x  = (const float*)d_in[0];
  const float* bwd_x  = (const float*)d_in[1];
  const float* w_in   = (const float*)d_in[2];   // (3E, E)
  const float* b_in   = (const float*)d_in[3];   // (3E,)
  const float* out_w  = (const float*)d_in[4];   // (E, E)
  const float* out_b  = (const float*)d_in[5];   // (E,)
  float* out = (float*)d_out;

  const size_t NQIN = (size_t)TT * BB * EE;  // 4M
  const size_t NKV  = 2 * NQIN;              // 8M
  const size_t NAVG = (size_t)BB * TT * SS;  // 8M

  // Workspace layout (f16): 40M elems = 80 MB total
  _Float16* ws16   = (_Float16*)d_ws;
  _Float16* qin_h  = ws16;
  _Float16* kv_h   = qin_h  + NQIN;
  _Float16* wqkv_h = kv_h   + NKV;
  _Float16* wo_h   = wqkv_h + (size_t)3 * EE * EE;
  _Float16* q_h    = wo_h   + (size_t)EE * EE;
  _Float16* k_h    = q_h    + NQIN;
  _Float16* vt_h   = k_h    + NKV;           // V transposed: [B][H][DH][SS], 8M
  _Float16* attn_h = vt_h   + NKV;

  float* avg_out = out + NQIN;  // second tuple element, flat-concatenated

  // 1) prep q_in / kv
  bmha_prep<<<dim3((unsigned)((NQIN + 255) / 256)), 256, 0, stream>>>(fwd_x, bwd_x, qin_h, kv_h);
  // 2) weights -> f16
  bmha_convw<<<dim3((unsigned)(((size_t)3 * EE * EE + 255) / 256)), 256, 0, stream>>>(
      w_in, out_w, wqkv_h, wo_h);
  // 3) zero avg-probs region (accumulated atomically by attention kernel)
  bmha_zero_f32<<<dim3((unsigned)((NAVG + 255) / 256)), 256, 0, stream>>>(avg_out, NAVG);

  // 4) QKV projections (C = A @ W^T + bias); V written transposed for PV matmul
  bmha_gemm_f16<<<dim3(EE / 256, (TT * BB) / 16), 128, 0, stream>>>(
      qin_h, wqkv_h, b_in, q_h, nullptr, nullptr, TT * BB, EE, EE);
  bmha_gemm_f16<<<dim3(EE / 256, (SS * BB) / 16), 128, 0, stream>>>(
      kv_h, wqkv_h + (size_t)EE * EE, b_in + EE, k_h, nullptr, nullptr, SS * BB, EE, EE);
  bmha_gemm_f16<<<dim3(EE / 256, (SS * BB) / 16), 128, 0, stream>>>(
      kv_h, wqkv_h + (size_t)2 * EE * EE, b_in + 2 * EE, nullptr, nullptr, vt_h,
      SS * BB, EE, EE);

  // 5) fused attention: 128 KB score panel + 16 KB TDM staging = 144 KB dynamic LDS
  const int lds_bytes = 16 * SS * (int)sizeof(float) + 4 * 2 * 2048;
  hipFuncSetAttribute(reinterpret_cast<const void*>(bmha_attn),
                      hipFuncAttributeMaxDynamicSharedMemorySize, lds_bytes);
  bmha_attn<<<dim3(TT / 16, HH, BB), 128, (size_t)lds_bytes, stream>>>(
      q_h, k_h, vt_h, avg_out, attn_h);

  // 6) output projection -> f32 into d_out[0 : T*B*E)
  bmha_gemm_f16<<<dim3(EE / 256, (TT * BB) / 16), 128, 0, stream>>>(
      attn_h, wo_h, out_b, nullptr, out, nullptr, TT * BB, EE, EE);
}
